// AmharicHNet_36567351558268
// MI455X (gfx1250) — matrix operand order repaired and checked by reading, not compiled, not run on hardware
//
#include <hip/hip_runtime.h>

#define Dm  768
#define FFd 2048
#define NH  12
#define HD  64
#define NB  2
#define SS  1024
#define SH  512
#define VOC 256

typedef __attribute__((ext_vector_type(16))) __bf16       v16bf;
typedef __attribute__((ext_vector_type(8)))  float        v8f;
typedef __attribute__((ext_vector_type(4)))  float        f32x4;
typedef __attribute__((ext_vector_type(4)))  unsigned int u32x4;

__device__ __forceinline__ unsigned int f2bf(float f) {
    unsigned int u = __float_as_uint(f);
    return (u + 0x7FFFu + ((u >> 16) & 1u)) >> 16;   // round-to-nearest-even bf16
}
__device__ __forceinline__ unsigned int pk2(float a, float b) {
    return f2bf(a) | (f2bf(b) << 16);
}

// ---------------------------------------------------------------------------
// Generic GEMM:  C[M,N] = scale * A[M,K] @ B(^T) + bias, optional ReLU.
//   transB=1: B stored [N,K] row-major (weights / K-matrix)   -> A @ B^T
//   transB=0: B stored [K,N] row-major (V-matrix)             -> A @ B
// Batched over gridDim.z: z -> (bz = z/H, hz = z%H), pointer offsets via strides.
// Block tile 128(M) x 64(N), K-tile 32. 8 waves, each wave owns a 32x32 tile
// (4 WMMA accumulators, 2 A-frags x 2 B-frags). Global->reg prefetch of the
// next K-tile overlaps the WMMA compute; global_prefetch_b8 pulls tile k+2.
// Requires: M % 128 == 0, N % 64 == 0, K % 32 == 0 (true for every call here).
// ---------------------------------------------------------------------------
__global__ __launch_bounds__(256)
void gemm_bf16_wmma(const float* __restrict__ A, const float* __restrict__ Bm,
                    float* __restrict__ C, const float* __restrict__ bias,
                    int K, int lda, int ldb, int ldc,
                    long sAb, long sAh, long sBb, long sBh, long sCb, long sCh,
                    int H, int transB, int relu, float scale)
{
    __shared__ unsigned short As[128 * 40];  // 128 rows x 32 K (stride 40, 16B-aligned rows)
    __shared__ unsigned short Bs[64 * 40];   // 64 N-rows x 32 K

    const int z  = blockIdx.z;
    const int bz = z / H;
    const int hz = z - bz * H;
    A  += (long)bz * sAb + (long)hz * sAh;
    Bm += (long)bz * sBb + (long)hz * sBh;
    C  += (long)bz * sCb + (long)hz * sCh;

    const int row0 = blockIdx.y << 7;        // 128-row tile
    const int col0 = blockIdx.x << 6;        // 64-col tile

    const int tid  = threadIdx.x;
    const int lane = tid & 31;
    const int wave = tid >> 5;
    const int wm   = wave >> 1;              // 0..3 -> rows wm*32
    const int wn   = wave & 1;               // 0..1 -> cols wn*32
    const int half = lane >> 4;
    const int l16  = lane & 15;

    // staging coordinates
    const int ar = tid >> 1,  ac = (tid & 1) << 4;   // A: 128x32, 16 f32/thread
    const int br = tid >> 2,  bc = (tid & 3) << 3;   // B(transB=1): 64x32, 8 f32/thread
    const int kr = tid >> 3,  nc = (tid & 7) << 3;   // B(transB=0): 32x64, 8 f32/thread

    f32x4 pa0, pa1, pa2, pa3, pb0, pb1;

    auto loadA = [&](int kk) {
        const float* g = A + (long)(row0 + ar) * lda + (kk + ac);
        pa0 = *(const f32x4*)(g);
        pa1 = *(const f32x4*)(g + 4);
        pa2 = *(const f32x4*)(g + 8);
        pa3 = *(const f32x4*)(g + 12);
    };
    auto loadB = [&](int kk) {
        if (transB) {
            const float* g = Bm + (long)(col0 + br) * ldb + (kk + bc);
            pb0 = *(const f32x4*)(g);
            pb1 = *(const f32x4*)(g + 4);
        } else {
            const float* g = Bm + (long)(kk + kr) * ldb + (col0 + nc);
            pb0 = *(const f32x4*)(g);
            pb1 = *(const f32x4*)(g + 4);
        }
    };
    auto stage = [&]() {
        u32x4 q;
        q.x = pk2(pa0.x, pa0.y); q.y = pk2(pa0.z, pa0.w);
        q.z = pk2(pa1.x, pa1.y); q.w = pk2(pa1.z, pa1.w);
        *(u32x4*)&As[ar * 40 + ac] = q;
        q.x = pk2(pa2.x, pa2.y); q.y = pk2(pa2.z, pa2.w);
        q.z = pk2(pa3.x, pa3.y); q.w = pk2(pa3.z, pa3.w);
        *(u32x4*)&As[ar * 40 + ac + 8] = q;
        if (transB) {
            q.x = pk2(pb0.x, pb0.y); q.y = pk2(pb0.z, pb0.w);
            q.z = pk2(pb1.x, pb1.y); q.w = pk2(pb1.z, pb1.w);
            *(u32x4*)&Bs[br * 40 + bc] = q;
        } else {                     // transpose while staging
            Bs[(nc + 0) * 40 + kr] = (unsigned short)f2bf(pb0.x);
            Bs[(nc + 1) * 40 + kr] = (unsigned short)f2bf(pb0.y);
            Bs[(nc + 2) * 40 + kr] = (unsigned short)f2bf(pb0.z);
            Bs[(nc + 3) * 40 + kr] = (unsigned short)f2bf(pb0.w);
            Bs[(nc + 4) * 40 + kr] = (unsigned short)f2bf(pb1.x);
            Bs[(nc + 5) * 40 + kr] = (unsigned short)f2bf(pb1.y);
            Bs[(nc + 6) * 40 + kr] = (unsigned short)f2bf(pb1.z);
            Bs[(nc + 7) * 40 + kr] = (unsigned short)f2bf(pb1.w);
        }
    };

    v8f acc00 = {0.f,0.f,0.f,0.f,0.f,0.f,0.f,0.f};
    v8f acc01 = acc00, acc10 = acc00, acc11 = acc00;

    loadA(0);
    loadB(0);

    for (int k0 = 0; k0 < K; k0 += 32) {
        __syncthreads();                 // previous compute done before LDS overwrite
        stage();
        __syncthreads();

        const int kn = k0 + 32;
        if (kn < K) {
            loadA(kn);                   // prefetch next K-tile into registers,
            loadB(kn);                   // overlapped with the WMMA compute below
            // pull tile k+2 toward the caches (global_prefetch_b8; speculative)
            __builtin_prefetch(A + (long)(row0 + ar) * lda + kn + 32, 0, 1);
            __builtin_prefetch(Bm + (transB ? (long)(col0 + br) * ldb + kn + 32
                                            : (long)(kn + 32 + kr) * ldb + col0 + nc), 0, 1);
        }

        union FragU { v16bf v; u32x4 q[2]; } fa0, fa1, fb0, fb1;
        // A fragment: lane(l16,half) -> row M, K = {half*8..+7, 16+half*8..+7}
        const unsigned short* pA0 = &As[(wm * 32 + l16) * 40 + (half << 3)];
        fa0.q[0] = *(const u32x4*)(pA0);
        fa0.q[1] = *(const u32x4*)(pA0 + 16);
        const unsigned short* pA1 = pA0 + 16 * 40;
        fa1.q[0] = *(const u32x4*)(pA1);
        fa1.q[1] = *(const u32x4*)(pA1 + 16);
        // B fragment: lane(l16,half) -> col N=l16, K = {half*8..+7, 16+half*8..+7}
        const unsigned short* pB0 = &Bs[(wn * 32 + l16) * 40 + (half << 3)];
        fb0.q[0] = *(const u32x4*)(pB0);
        fb0.q[1] = *(const u32x4*)(pB0 + 16);
        const unsigned short* pB1 = pB0 + 16 * 40;
        fb1.q[0] = *(const u32x4*)(pB1);
        fb1.q[1] = *(const u32x4*)(pB1 + 16);

        acc00 = __builtin_amdgcn_wmma_f32_16x16x32_bf16(false, fa0.v, false, fb0.v,
                                                        (short)0, acc00, false, false);
        acc01 = __builtin_amdgcn_wmma_f32_16x16x32_bf16(false, fa0.v, false, fb1.v,
                                                        (short)0, acc01, false, false);
        acc10 = __builtin_amdgcn_wmma_f32_16x16x32_bf16(false, fa1.v, false, fb0.v,
                                                        (short)0, acc10, false, false);
        acc11 = __builtin_amdgcn_wmma_f32_16x16x32_bf16(false, fa1.v, false, fb1.v,
                                                        (short)0, acc11, false, false);
    }

    // epilogue: C layout — lane(l16,half), reg r -> (M = r + 8*half, N = l16)
    const int gn0 = col0 + wn * 32 + l16;
    const int gn1 = gn0 + 16;
    const float b0 = bias ? bias[gn0] : 0.f;
    const float b1 = bias ? bias[gn1] : 0.f;
#pragma unroll
    for (int r = 0; r < 8; r++) {
        int gmA = row0 + wm * 32 + (half << 3) + r;
        int gmB = gmA + 16;
        float v00 = acc00[r] * scale + b0;
        float v01 = acc01[r] * scale + b1;
        float v10 = acc10[r] * scale + b0;
        float v11 = acc11[r] * scale + b1;
        if (relu) {
            v00 = v00 > 0.f ? v00 : 0.f;  v01 = v01 > 0.f ? v01 : 0.f;
            v10 = v10 > 0.f ? v10 : 0.f;  v11 = v11 > 0.f ? v11 : 0.f;
        }
        C[(long)gmA * ldc + gn0] = v00;
        C[(long)gmA * ldc + gn1] = v01;
        C[(long)gmB * ldc + gn0] = v10;
        C[(long)gmB * ldc + gn1] = v11;
    }
}

// ---------------------------------------------------------------------------
// Embedding + sinusoidal positional encoding
// ---------------------------------------------------------------------------
__global__ void embed_pe_kernel(const int* __restrict__ ids, const float* __restrict__ emb,
                                float* __restrict__ X, int total)
{
    int i = blockIdx.x * 256 + threadIdx.x;
    if (i >= total) return;
    int t   = i / Dm;
    int j   = i - t * Dm;
    int pos = t & (SS - 1);
    int hi  = j >> 1;
    float freq = __expf(-(float)(2 * hi) * (9.210340371976184f / (float)Dm));
    float ang  = (float)pos * freq;
    float pe   = (j & 1) ? __cosf(ang) : __sinf(ang);
    X[i] = emb[(long)ids[t] * Dm + j] + pe;
}

// ---------------------------------------------------------------------------
// Boundary probs: sigmoid(x . w + b), one block per token
// ---------------------------------------------------------------------------
__global__ __launch_bounds__(256)
void boundary_kernel(const float* __restrict__ X, const float* __restrict__ w,
                     const float* __restrict__ bsc, float* __restrict__ probs)
{
    __shared__ float red[256];
    int t = blockIdx.x, tid = threadIdx.x;
    float s = 0.f;
    for (int j = tid; j < Dm; j += 256) s += X[(long)t * Dm + j] * w[j];
    red[tid] = s; __syncthreads();
    for (int o = 128; o > 0; o >>= 1) { if (tid < o) red[tid] += red[tid + o]; __syncthreads(); }
    if (tid == 0) probs[t] = 1.f / (1.f + __expf(-(red[0] + bsc[0])));
}

// ---------------------------------------------------------------------------
// out = LN(x (+ r)) * w + b   (D = 768, one block per token, 3 elems/thread)
// ---------------------------------------------------------------------------
__global__ __launch_bounds__(256)
void ln_res_kernel(const float* __restrict__ x, const float* __restrict__ r,
                   float* __restrict__ out, const float* __restrict__ w,
                   const float* __restrict__ b)
{
    __shared__ float red[256];
    int t = blockIdx.x, tid = threadIdx.x;
    long base = (long)t * Dm;
    float v0 = x[base + tid      ] + (r ? r[base + tid      ] : 0.f);
    float v1 = x[base + tid + 256] + (r ? r[base + tid + 256] : 0.f);
    float v2 = x[base + tid + 512] + (r ? r[base + tid + 512] : 0.f);
    red[tid] = v0 + v1 + v2; __syncthreads();
    for (int o = 128; o > 0; o >>= 1) { if (tid < o) red[tid] += red[tid + o]; __syncthreads(); }
    float mean = red[0] * (1.f / 768.f);
    __syncthreads();
    float d0 = v0 - mean, d1 = v1 - mean, d2 = v2 - mean;
    red[tid] = d0 * d0 + d1 * d1 + d2 * d2; __syncthreads();
    for (int o = 128; o > 0; o >>= 1) { if (tid < o) red[tid] += red[tid + o]; __syncthreads(); }
    float rstd = rsqrtf(red[0] * (1.f / 768.f) + 1e-5f);
    out[base + tid      ] = d0 * rstd * w[tid      ] + b[tid      ];
    out[base + tid + 256] = d1 * rstd * w[tid + 256] + b[tid + 256];
    out[base + tid + 512] = d2 * rstd * w[tid + 512] + b[tid + 512];
}

// ---------------------------------------------------------------------------
// Row softmax (in-place), one block per row
// ---------------------------------------------------------------------------
__global__ __launch_bounds__(256)
void softmax_kernel(float* __restrict__ S, int Sk)
{
    __shared__ float red[256];
    long base = (long)blockIdx.x * Sk;
    int tid = threadIdx.x;
    float m = -3.4e38f;
    for (int c = tid; c < Sk; c += 256) { float v = S[base + c]; m = v > m ? v : m; }
    red[tid] = m; __syncthreads();
    for (int o = 128; o > 0; o >>= 1) {
        if (tid < o) red[tid] = red[tid] > red[tid + o] ? red[tid] : red[tid + o];
        __syncthreads();
    }
    m = red[0]; __syncthreads();
    float s = 0.f;
    for (int c = tid; c < Sk; c += 256) { float e = __expf(S[base + c] - m); S[base + c] = e; s += e; }
    red[tid] = s; __syncthreads();
    for (int o = 128; o > 0; o >>= 1) { if (tid < o) red[tid] += red[tid + o]; __syncthreads(); }
    float inv = 1.f / red[0];
    for (int c = tid; c < Sk; c += 256) S[base + c] *= inv;
}

// ---------------------------------------------------------------------------
// Dynamic chunker: segment means per boundary cumsum, one block per batch
// ---------------------------------------------------------------------------
__global__ __launch_bounds__(256)
void chunk_kernel(const float* __restrict__ X, const float* __restrict__ probs,
                  float* __restrict__ out)
{
    __shared__ int seg[SS];
    __shared__ int cstart[SS];
    __shared__ int ccnt[SS];
    int b = blockIdx.x, tid = threadIdx.x;
    if (tid == 0) {
        int cur = 0;
        for (int i = 0; i < SS; i++) {
            seg[i] = cur;                               // exclusive prefix of boundaries
            bool e = (i == SS - 1) ? true
                   : (i == 0)      ? false
                   : (probs[b * SS + i] > 0.5f);
            if (e) cur++;
        }
        for (int c = 0; c < SS; c++) { cstart[c] = 0; ccnt[c] = 0; }
        for (int i = 0; i < SS; i++) {
            int c = seg[i];
            if (ccnt[c] == 0) cstart[c] = i;
            ccnt[c]++;
        }
    }
    __syncthreads();
    for (int idx = tid; idx < SH * Dm; idx += 256) {
        int c = idx / Dm, j = idx - c * Dm;
        int cnt = ccnt[c], st = cstart[c];
        float s = 0.f;
        for (int i = 0; i < cnt; i++) s += X[((long)b * SS + st + i) * Dm + j];
        out[((long)b * SH + c) * Dm + j] = cnt ? s / (float)cnt : 0.f;
    }
}

// ===========================================================================
// Host orchestration
// ===========================================================================
struct Enc {
    const float *Wq,*Wk,*Wv,*Wo,*bq,*bk,*bv,*bo,*W1,*b1,*W2,*b2,*l1w,*l1b,*l2w,*l2b;
};
static Enc grab(void* const* d, int base) {
    Enc e;
    e.Wq  = (const float*)d[base + 0];  e.Wk  = (const float*)d[base + 1];
    e.Wv  = (const float*)d[base + 2];  e.Wo  = (const float*)d[base + 3];
    e.bq  = (const float*)d[base + 4];  e.bk  = (const float*)d[base + 5];
    e.bv  = (const float*)d[base + 6];  e.bo  = (const float*)d[base + 7];
    e.W1  = (const float*)d[base + 8];  e.b1  = (const float*)d[base + 9];
    e.W2  = (const float*)d[base +10];  e.b2  = (const float*)d[base +11];
    e.l1w = (const float*)d[base +12];  e.l1b = (const float*)d[base +13];
    e.l2w = (const float*)d[base +14];  e.l2b = (const float*)d[base +15];
    return e;
}

static void gemm(hipStream_t st, const float* A, const float* Bm, float* C, const float* bias,
                 int M, int N, int K, int lda, int ldb, int ldc,
                 long sAb, long sAh, long sBb, long sBh, long sCb, long sCh,
                 int nb, int H, int transB, int relu, float scale)
{
    dim3 g((unsigned)(N / 64), (unsigned)(M / 128), (unsigned)nb);
    gemm_bf16_wmma<<<g, 256, 0, st>>>(A, Bm, C, bias, K, lda, ldb, ldc,
                                      sAb, sAh, sBb, sBh, sCb, sCh,
                                      H, transB, relu, scale);
}

static void mha(hipStream_t st, const float* Xq, const float* Xkv,
                float* concatBuf, float* projOut,
                const float* Wq, const float* bq, const float* Wk, const float* bk,
                const float* Wv, const float* bv, const float* Wo, const float* bo,
                int Sq, int Sk, float* Q, float* Kb, float* V, float* SC)
{
    int Mq = NB * Sq, Mk = NB * Sk;
    gemm(st, Xq,  Wq, Q,  bq, Mq, Dm, Dm, Dm, Dm, Dm, 0,0,0,0,0,0, 1,1, 1,0, 1.f);
    gemm(st, Xkv, Wk, Kb, bk, Mk, Dm, Dm, Dm, Dm, Dm, 0,0,0,0,0,0, 1,1, 1,0, 1.f);
    gemm(st, Xkv, Wv, V,  bv, Mk, Dm, Dm, Dm, Dm, Dm, 0,0,0,0,0,0, 1,1, 1,0, 1.f);
    // scores[b,h] = (Q_slice @ K_slice^T) / sqrt(64)
    gemm(st, Q, Kb, SC, nullptr, Sq, Sk, HD, Dm, Dm, Sk,
         (long)Sq * Dm, HD, (long)Sk * Dm, HD, (long)NH * Sq * Sk, (long)Sq * Sk,
         NB * NH, NH, 1, 0, 0.125f);
    softmax_kernel<<<NB * NH * Sq, 256, 0, st>>>(SC, Sk);
    // out[b,h] = P @ V_slice, written into concat head slots
    gemm(st, SC, V, concatBuf, nullptr, Sq, HD, Sk, Sk, Dm, Dm,
         (long)NH * Sq * Sk, (long)Sq * Sk, (long)Sk * Dm, HD, (long)Sq * Dm, HD,
         NB * NH, NH, 0, 0, 1.f);
    gemm(st, concatBuf, Wo, projOut, bo, Mq, Dm, Dm, Dm, Dm, Dm, 0,0,0,0,0,0, 1,1, 1,0, 1.f);
}

static void enc_layer(hipStream_t st, float* Xb, const Enc& e, int l, int Sq,
                      float* Q, float* Kb, float* V, float* BR1, float* BR2,
                      float* FFB, float* SC)
{
    size_t wo = (size_t)l * Dm * Dm;
    mha(st, Xb, Xb, BR1, BR2,
        e.Wq + wo, e.bq + (size_t)l * Dm, e.Wk + wo, e.bk + (size_t)l * Dm,
        e.Wv + wo, e.bv + (size_t)l * Dm, e.Wo + wo, e.bo + (size_t)l * Dm,
        Sq, Sq, Q, Kb, V, SC);
    ln_res_kernel<<<NB * Sq, 256, 0, st>>>(Xb, BR2, Xb, e.l1w + (size_t)l * Dm, e.l1b + (size_t)l * Dm);
    gemm(st, Xb, e.W1 + (size_t)l * FFd * Dm, FFB, e.b1 + (size_t)l * FFd,
         NB * Sq, FFd, Dm, Dm, Dm, FFd, 0,0,0,0,0,0, 1,1, 1,1, 1.f);
    gemm(st, FFB, e.W2 + (size_t)l * Dm * FFd, BR2, e.b2 + (size_t)l * Dm,
         NB * Sq, Dm, FFd, FFd, FFd, Dm, 0,0,0,0,0,0, 1,1, 1,0, 1.f);
    ln_res_kernel<<<NB * Sq, 256, 0, st>>>(Xb, BR2, Xb, e.l2w + (size_t)l * Dm, e.l2b + (size_t)l * Dm);
}

extern "C" void kernel_launch(void* const* d_in, const int* in_sizes, int n_in,
                              void* d_out, int out_size, void* d_ws, size_t ws_size,
                              hipStream_t stream)
{
    (void)in_sizes; (void)n_in; (void)out_size; (void)ws_size;

    const int*   input_ids  = (const int*)d_in[0];
    const int*   target_ids = (const int*)d_in[1];
    const float* emb        = (const float*)d_in[2];
    const float* chunk_w    = (const float*)d_in[3];
    const float* chunk_b    = (const float*)d_in[4];
    Enc be = grab(d_in, 5);     // byte_enc  (L=2)
    Enc ce = grab(d_in, 21);    // chunk_enc (L=2)
    Enc me = grab(d_in, 37);    // main_enc  (L=12)
    Enc de = grab(d_in, 53);    // dec base  (L=4)
    const float* Wq2  = (const float*)d_in[69];
    const float* Wk2  = (const float*)d_in[70];
    const float* Wv2  = (const float*)d_in[71];
    const float* Wo2  = (const float*)d_in[72];
    const float* bq2  = (const float*)d_in[73];
    const float* bk2  = (const float*)d_in[74];
    const float* bv2  = (const float*)d_in[75];
    const float* bo2  = (const float*)d_in[76];
    const float* lncw = (const float*)d_in[77];
    const float* lncb = (const float*)d_in[78];
    const float* ln_w = (const float*)d_in[79];
    const float* ln_b = (const float*)d_in[80];
    const float* outw = (const float*)d_in[81];
    const float* outb = (const float*)d_in[82];

    float* logits = (float*)d_out;                       // [B,S,VOC]
    float* probs  = logits + (long)NB * SS * VOC;        // [B,S]

    float* W = (float*)d_ws;
    size_t o = 0;
    float* X   = W + o; o += (size_t)NB * SS * Dm;       // byte-level stream
    float* XD  = W + o; o += (size_t)NB * SS * Dm;       // decoder stream
    float* MEM = W + o; o += (size_t)NB * SH * Dm;       // chunk-level stream / memory
    float* Q   = W + o; o += (size_t)NB * SS * Dm;
    float* Kb  = W + o; o += (size_t)NB * SS * Dm;
    float* V   = W + o; o += (size_t)NB * SS * Dm;
    float* BR1 = W + o; o += (size_t)NB * SS * Dm;       // head-concat buffer
    float* BR2 = W + o; o += (size_t)NB * SS * Dm;       // branch (pre-LN) buffer
    float* FFB = W + o; o += (size_t)NB * SS * FFd;
    float* SC  = W + o; o += (size_t)NB * NH * SS * SS;  // attention scores

    const int totX = NB * SS * Dm;

    // 1. source embedding + PE, boundary probs (straight into d_out tail)
    embed_pe_kernel<<<(totX + 255) / 256, 256, 0, stream>>>(input_ids, emb, X, totX);
    boundary_kernel<<<NB * SS, 256, 0, stream>>>(X, chunk_w, chunk_b, probs);

    // 2. byte encoder (2 layers, Sq=1024)
    for (int l = 0; l < 2; l++) enc_layer(stream, X, be, l, SS, Q, Kb, V, BR1, BR2, FFB, SC);

    // 3. dynamic chunking -> [B, 512, D]
    chunk_kernel<<<NB, 256, 0, stream>>>(X, probs, MEM);

    // 4. chunk encoder (2) + main encoder (12), Sq=512
    for (int l = 0; l < 2;  l++) enc_layer(stream, MEM, ce, l, SH, Q, Kb, V, BR1, BR2, FFB, SC);
    for (int l = 0; l < 12; l++) enc_layer(stream, MEM, me, l, SH, Q, Kb, V, BR1, BR2, FFB, SC);

    // 5. decoder (4 layers): self-attn + cross-attn(memory) + FF
    embed_pe_kernel<<<(totX + 255) / 256, 256, 0, stream>>>(target_ids, emb, XD, totX);
    for (int l = 0; l < 4; l++) {
        size_t wo = (size_t)l * Dm * Dm, bo_ = (size_t)l * Dm;
        mha(stream, XD, XD, BR1, BR2,
            de.Wq + wo, de.bq + bo_, de.Wk + wo, de.bk + bo_,
            de.Wv + wo, de.bv + bo_, de.Wo + wo, de.bo + bo_,
            SS, SS, Q, Kb, V, SC);
        ln_res_kernel<<<NB * SS, 256, 0, stream>>>(XD, BR2, XD, de.l1w + bo_, de.l1b + bo_);
        mha(stream, XD, MEM, BR1, BR2,
            Wq2 + wo, bq2 + bo_, Wk2 + wo, bk2 + bo_,
            Wv2 + wo, bv2 + bo_, Wo2 + wo, bo2 + bo_,
            SS, SH, Q, Kb, V, SC);
        ln_res_kernel<<<NB * SS, 256, 0, stream>>>(XD, BR2, XD, lncw + bo_, lncb + bo_);
        gemm(stream, XD, de.W1 + (size_t)l * FFd * Dm, FFB, de.b1 + (size_t)l * FFd,
             NB * SS, FFd, Dm, Dm, Dm, FFd, 0,0,0,0,0,0, 1,1, 1,1, 1.f);
        gemm(stream, FFB, de.W2 + (size_t)l * Dm * FFd, BR2, de.b2 + bo_,
             NB * SS, Dm, FFd, FFd, FFd, Dm, 0,0,0,0,0,0, 1,1, 1,0, 1.f);
        ln_res_kernel<<<NB * SS, 256, 0, stream>>>(XD, BR2, XD, de.l2w + bo_, de.l2b + bo_);
    }

    // 6. final LN + vocab projection
    ln_res_kernel<<<NB * SS, 256, 0, stream>>>(XD, nullptr, XD, ln_w, ln_b);
    gemm(stream, XD, outw, logits, outb, NB * SS, VOC, Dm, Dm, Dm, VOC,
         0,0,0,0,0,0, 1,1, 1,0, 1.f);
}
